// SwinTransformerBlock_16707422781479
// MI455X (gfx1250) — compile-verified, matrix operand order checked by reading, not run on hardware
//
#include <hip/hip_runtime.h>
#include <hip/hip_bf16.h>

// ---------------------------------------------------------------------------
// Swin block, MI455X (gfx1250, wave32). All GEMMs on v_wmma_f32_16x16x32_f16.
// Weights pre-transposed to [N][K] f16 so every LDS tile is a contiguous
// 16B-chunk block copy done with global_load_async_to_lds_b128 (ASYNCcnt),
// double-buffered so tile k+1 loads overlap WMMA on tile k. One barrier per
// K-step (the next iteration's barrier covers the WAR on the buffer).
// ---------------------------------------------------------------------------

typedef _Float16 f16x8  __attribute__((ext_vector_type(8)));
typedef _Float16 f16x16 __attribute__((ext_vector_type(16)));
typedef float    f32x8  __attribute__((ext_vector_type(8)));

#define WS_ 7
#define SHIFT_ 3
#define NH_ 12
#define C_ 384
#define HD_ 32
#define NTOK 100352     // 32 * 64 * 49
#define NWIN 2048       // 32 * 64

__device__ __forceinline__ f32x8 wmma32(f16x16 a, f16x16 b, f32x8 c) {
  return __builtin_amdgcn_wmma_f32_16x16x32_f16(false, a, false, b, (short)0, c,
                                                false, false);
}

// CDNA5 async copy: 16B global -> LDS, tracked by ASYNCcnt.
__device__ __forceinline__ void async_copy16(const _Float16* lds,
                                             const _Float16* gptr) {
  unsigned lo = (unsigned)(uintptr_t)lds;
  unsigned long long ga = (unsigned long long)(uintptr_t)gptr;
  asm volatile("global_load_async_to_lds_b128 %0, %1, off"
               :: "v"(lo), "v"(ga) : "memory");
}
__device__ __forceinline__ void wait_async0() {
  asm volatile("s_wait_asynccnt 0x0" ::: "memory");
}

// Load one 16x32 f16 A/B^T fragment from LDS (row-major, ld halves, ld%8==0).
// ISA 7.12.2: lanes 0-15 rows 0-15 K{0..7,16..23}; lanes 16-31 K shifted +8.
__device__ __forceinline__ f16x16 load_frag(const _Float16* base, int ld) {
  int lane = threadIdx.x & 31;
  int r  = lane & 15;
  int hb = (lane >> 4) << 3;
  const _Float16* p = base + r * ld + hb;
  f16x8 lo = *(const f16x8*)(p);
  f16x8 hi = *(const f16x8*)(p + 16);
  f16x16 f;
#pragma unroll
  for (int i = 0; i < 8; i++) { f[i] = lo[i]; f[i + 8] = hi[i]; }
  return f;
}

// window-token index t -> flat spatial token p (applies the -SHIFT roll).
__device__ __forceinline__ int token_to_pos(int t) {
  int win = t / 49, n = t - win * 49;
  int b = win >> 6, w = win & 63;
  int wy = w >> 3, wx = w & 7;
  int iy = n / 7, ix = n - iy * 7;
  int r = wy * 7 + iy + SHIFT_; if (r >= 56) r -= 56;
  int c = wx * 7 + ix + SHIFT_; if (c >= 56) c -= 56;
  return b * 3136 + r * 56 + c;
}

// ---------------------------------------------------------------------------
// f32 [K][N] -> f16 transposed [N][K]
__global__ __launch_bounds__(256) void cvt_transpose_kernel(
    const float* __restrict__ in, _Float16* __restrict__ out, int K, int N) {
  int i = blockIdx.x * 256 + threadIdx.x;
  if (i < K * N) {
    int k = i / N, n = i - k * N;
    out[(size_t)n * K + k] = (_Float16)in[i];
  }
}

// LayerNorm over C=384; one wave per token. mode 0: LN1 + shift + partition.
__global__ __launch_bounds__(256) void ln_kernel(const float* __restrict__ xin,
                                                 const float* __restrict__ g,
                                                 const float* __restrict__ bb,
                                                 _Float16* __restrict__ outH,
                                                 int mode) {
  int wid = threadIdx.x >> 5, lane = threadIdx.x & 31;
  int t = blockIdx.x * 8 + wid;
  int p = (mode == 0) ? token_to_pos(t) : t;
  const float* xr = xin + (size_t)p * C_;
  float vals[12];
  float s = 0.f;
#pragma unroll
  for (int i = 0; i < 12; i++) { float v = xr[lane + i * 32]; vals[i] = v; s += v; }
#pragma unroll
  for (int m = 16; m >= 1; m >>= 1) s += __shfl_xor(s, m, 32);
  float mu = s * (1.0f / 384.0f);
  float s2 = 0.f;
#pragma unroll
  for (int i = 0; i < 12; i++) { float d = vals[i] - mu; s2 += d * d; }
#pragma unroll
  for (int m = 16; m >= 1; m >>= 1) s2 += __shfl_xor(s2, m, 32);
  float rstd = rsqrtf(s2 * (1.0f / 384.0f) + 1e-5f);
  _Float16* orow = outH + (size_t)t * C_;
#pragma unroll
  for (int i = 0; i < 12; i++) {
    int c = lane + i * 32;
    orow[c] = (_Float16)((vals[i] - mu) * rstd * g[c] + bb[c]);
  }
}

// ---------------------------------------------------------------------------
// Tiled WMMA GEMM: C[M,N] = A[M,K](f16) * Bt[N,K](f16, pre-transposed).
// Block 256 thr (8 waves), tile 128x128, BK=32, double-buffered LDS filled by
// async-to-LDS. Wave (wm,wn) owns 32 rows x 64 cols = 2x4 WMMA tiles.
// EPI 0: QKV scatter  1: proj+unshift+residual  2: GELU f16  3: residual out
template <int EPI>
__global__ __launch_bounds__(256) void gemm_wmma(
    const _Float16* __restrict__ A, const _Float16* __restrict__ Bt_g,
    const float* __restrict__ bias, const float* __restrict__ res,
    float* __restrict__ outF, _Float16* __restrict__ outH,
    _Float16* __restrict__ outH2, _Float16* __restrict__ outH3,
    int M, int N, int K) {
  __shared__ alignas(16) _Float16 As[2][128][40];
  __shared__ alignas(16) _Float16 Bs[2][128][40];

  int tid = threadIdx.x;
  int m0 = blockIdx.y * 128;
  int n0 = blockIdx.x * 128;
  int wave = tid >> 5, lane = tid & 31;
  int wm = wave & 3, wn = wave >> 2;   // 4 row-strips x 2 col-strips

  f32x8 acc[2][4];
#pragma unroll
  for (int i = 0; i < 2; i++)
#pragma unroll
    for (int j = 0; j < 4; j++) acc[i][j] = (f32x8){};

  // 128 rows x 4 chunks(16B) per tile = 512 chunks; 2 per thread per matrix.
  auto load_tiles = [&](int buf, int k0) {
#pragma unroll
    for (int i = 0; i < 2; i++) {
      int e = tid + i * 256;
      int r = e >> 2, ck = (e & 3) * 8;
      async_copy16(&As[buf][r][ck], A + (size_t)(m0 + r) * K + k0 + ck);
      async_copy16(&Bs[buf][r][ck], Bt_g + (size_t)(n0 + r) * K + k0 + ck);
    }
  };

  int nk = K >> 5;
  load_tiles(0, 0);
  for (int kt = 0; kt < nk; kt++) {
    int buf = kt & 1;
    wait_async0();
    __syncthreads();
    if (kt + 1 < nk) load_tiles(buf ^ 1, (kt + 1) << 5);
    f16x16 a0 = load_frag(&As[buf][wm * 32][0], 40);
    f16x16 a1 = load_frag(&As[buf][wm * 32 + 16][0], 40);
#pragma unroll
    for (int j = 0; j < 4; j++) {
      f16x16 b = load_frag(&Bs[buf][wn * 64 + j * 16][0], 40);
      acc[0][j] = wmma32(a0, b, acc[0][j]);
      acc[1][j] = wmma32(a1, b, acc[1][j]);
    }
  }

  int lr = lane & 15, hi = lane >> 4;
#pragma unroll
  for (int i = 0; i < 2; i++)
#pragma unroll
    for (int j = 0; j < 4; j++)
#pragma unroll
      for (int v = 0; v < 8; v++) {
        int m = m0 + wm * 32 + i * 16 + v + hi * 8;
        int n = n0 + wn * 64 + j * 16 + lr;
        float val = acc[i][j][v] + bias[n];
        if (EPI == 0) {
          int which = n / C_; int r = n - which * C_;
          int head = r >> 5, d = r & 31;
          int win = m / 49, nn = m - win * 49;
          size_t bh = (size_t)win * NH_ + head;
          if (which == 0)
            outH[(bh * 64 + nn) * 32 + d] = (_Float16)(val * 0.1767766953f);
          else if (which == 1)
            outH2[(bh * 64 + nn) * 32 + d] = (_Float16)val;
          else
            outH3[(bh * 32 + d) * 64 + nn] = (_Float16)val;
        } else if (EPI == 1) {
          int p = token_to_pos(m);
          size_t idx = (size_t)p * C_ + n;
          outF[idx] = res[idx] + val;
        } else if (EPI == 2) {
          float ge = 0.5f * val * (1.0f + erff(val * 0.70710678f));
          outH[(size_t)m * N + n] = (_Float16)ge;
        } else {
          size_t idx = (size_t)m * C_ + n;
          outF[idx] = res[idx] + val;
        }
      }
}

// ---------------------------------------------------------------------------
// Attention: one block per (window, head). 4 waves. q,k: 64x32 f16 (rows>=49
// padded), vt: 32x64 (V transposed). S = q k^T (WMMA), softmax w/ rel-pos bias
// + analytic shift mask, P (f16) @ V (WMMA, K=64 in 2 steps) -> o token-major.
__global__ __launch_bounds__(128) void attn_kernel(
    const _Float16* __restrict__ q16, const _Float16* __restrict__ k16,
    const _Float16* __restrict__ vt16, const float* __restrict__ bias_table,
    _Float16* __restrict__ o16) {
  __shared__ alignas(16) _Float16 qs[64][40];
  __shared__ alignas(16) _Float16 ks[64][40];
  __shared__ alignas(16) _Float16 vts[32][72];
  __shared__ alignas(16) float    Ss[64][68];
  __shared__ alignas(16) _Float16 Ps[64][72];

  int tid = threadIdx.x;
  int wh = blockIdx.x;
  int win = wh / NH_, head = wh - win * NH_;
  size_t base = (size_t)wh * 2048;

  // Async tile fills: q,k 64rows x 4 chunks; vt 32rows x 8 chunks (256 each).
#pragma unroll
  for (int i = 0; i < 2; i++) {
    int e = tid + i * 128;
    int r = e >> 2, ck = (e & 3) * 8;
    async_copy16(&qs[r][ck], q16 + base + r * 32 + ck);
    async_copy16(&ks[r][ck], k16 + base + r * 32 + ck);
    int vr = e >> 3, vc = (e & 7) * 8;
    async_copy16(&vts[vr][vc], vt16 + base + vr * 64 + vc);
  }
  wait_async0();
  __syncthreads();

  int wave = tid >> 5, lane = tid & 31, lr = lane & 15, hi = lane >> 4;

  {  // S = q k^T : each wave one 16-row strip x 64 cols
    f16x16 a = load_frag(&qs[wave * 16][0], 40);
#pragma unroll
    for (int j = 0; j < 4; j++) {
      f16x16 b = load_frag(&ks[j * 16][0], 40);
      f32x8 c = (f32x8){};
      c = wmma32(a, b, c);
#pragma unroll
      for (int v = 0; v < 8; v++)
        Ss[wave * 16 + v + hi * 8][j * 16 + lr] = c[v];
    }
  }
  __syncthreads();

  if (tid < 64) {  // softmax, one thread per row
    int m = tid;
    if (m < 49) {
      int my = m / 7, mx = m - my * 7;
      int wi = win & 63, wy = wi >> 3, wx = wi & 7;
      int ay = wy * 7 + my, ax = wx * 7 + mx;
      int cm = (ay < 49 ? 0 : (ay < 53 ? 1 : 2)) * 3 +
               (ax < 49 ? 0 : (ax < 53 ? 1 : 2));
      float mx_ = -1e30f;
      for (int n = 0; n < 49; n++) {
        int ny = n / 7, nx = n - ny * 7;
        int idx = (my - ny + 6) * 13 + (mx - nx + 6);
        float rb = bias_table[idx * NH_ + head];
        int by = wy * 7 + ny, bx = wx * 7 + nx;
        int cn = (by < 49 ? 0 : (by < 53 ? 1 : 2)) * 3 +
                 (bx < 49 ? 0 : (bx < 53 ? 1 : 2));
        float sv = Ss[m][n] + rb + ((cn != cm) ? -100.0f : 0.0f);
        Ss[m][n] = sv;
        mx_ = fmaxf(mx_, sv);
      }
      float sum = 0.f;
      for (int n = 0; n < 49; n++) {
        float e = __expf(Ss[m][n] - mx_);
        sum += e;
        Ps[m][n] = (_Float16)e;
      }
      float inv = 1.0f / sum;
      for (int n = 0; n < 49; n++)
        Ps[m][n] = (_Float16)((float)Ps[m][n] * inv);
      for (int n = 49; n < 64; n++) Ps[m][n] = (_Float16)0.f;
    } else {
      for (int n = 0; n < 64; n++) Ps[m][n] = (_Float16)0.f;
    }
  }
  __syncthreads();

  {  // o = P V : per wave 16 rows x 32 dims (2 col tiles), K=64 in 2 steps
    f32x8 o0 = (f32x8){}, o1 = (f32x8){};
#pragma unroll
    for (int kt = 0; kt < 64; kt += 32) {
      f16x16 a  = load_frag(&Ps[wave * 16][kt], 72);
      f16x16 b0 = load_frag(&vts[0][kt], 72);
      f16x16 b1 = load_frag(&vts[16][kt], 72);
      o0 = wmma32(a, b0, o0);
      o1 = wmma32(a, b1, o1);
    }
#pragma unroll
    for (int v = 0; v < 8; v++) {
      int m = wave * 16 + v + hi * 8;
      if (m < 49) {
        size_t ob = ((size_t)win * 49 + m) * C_ + head * HD_;
        o16[ob + lr]      = (_Float16)o0[v];
        o16[ob + 16 + lr] = (_Float16)o1[v];
      }
    }
  }
}

// ---------------------------------------------------------------------------
extern "C" void kernel_launch(void* const* d_in, const int* in_sizes, int n_in,
                              void* d_out, int out_size, void* d_ws,
                              size_t ws_size, hipStream_t stream) {
  const float* x          = (const float*)d_in[0];
  const float* ln1_g      = (const float*)d_in[1];
  const float* ln1_b      = (const float*)d_in[2];
  const float* qkv_w      = (const float*)d_in[3];
  const float* qkv_b      = (const float*)d_in[4];
  const float* proj_w     = (const float*)d_in[5];
  const float* proj_b     = (const float*)d_in[6];
  const float* bias_table = (const float*)d_in[7];
  const float* ln2_g      = (const float*)d_in[8];
  const float* ln2_b      = (const float*)d_in[9];
  const float* fc1_w      = (const float*)d_in[10];
  const float* fc1_b      = (const float*)d_in[11];
  const float* fc2_w      = (const float*)d_in[12];
  const float* fc2_b      = (const float*)d_in[13];
  float* out = (float*)d_out;
  char* ws = (char*)d_ws;

  // ---- workspace arena (all offsets 256B aligned) ----
  size_t off = 0;
  auto take = [&](size_t bytes) {
    size_t o = off;
    off += (bytes + 255) & ~(size_t)255;
    return o;
  };
  _Float16* wq16 = (_Float16*)(ws + take((size_t)C_ * 3 * C_ * 2));  // [1152][384]
  _Float16* wp16 = (_Float16*)(ws + take((size_t)C_ * C_ * 2));      // [384][384]
  _Float16* wf1  = (_Float16*)(ws + take((size_t)C_ * 4 * C_ * 2));  // [1536][384]
  _Float16* wf2  = (_Float16*)(ws + take((size_t)4 * C_ * C_ * 2));  // [384][1536]
  size_t qkvOne = (size_t)NWIN * NH_ * 64 * 32 * 2;                  // 100.66MB
  size_t gBytes = (size_t)NTOK * 4 * C_ * 2;                         // 308.3MB
  size_t regA = take(gBytes > 3 * qkvOne ? gBytes : 3 * qkvOne);
  _Float16* q16  = (_Float16*)(ws + regA);
  _Float16* k16  = (_Float16*)(ws + regA + qkvOne);
  _Float16* vt16 = (_Float16*)(ws + regA + 2 * qkvOne);
  _Float16* g16  = (_Float16*)(ws + regA);          // reuse after attention
  size_t regB = take((size_t)NTOK * C_ * 2);        // h16 / o16 / h2 (77MB)
  _Float16* h16 = (_Float16*)(ws + regB);
  _Float16* o16 = (_Float16*)(ws + regB);
  _Float16* h2  = (_Float16*)(ws + regB);
  float* x2 = (float*)(ws + take((size_t)NTOK * C_ * 4));            // 154MB

  // 1) weight conversion f32 [K][N] -> f16 transposed [N][K]
  cvt_transpose_kernel<<<(442368 + 255) / 256, 256, 0, stream>>>(qkv_w, wq16, C_, 3 * C_);
  cvt_transpose_kernel<<<(147456 + 255) / 256, 256, 0, stream>>>(proj_w, wp16, C_, C_);
  cvt_transpose_kernel<<<(589824 + 255) / 256, 256, 0, stream>>>(fc1_w, wf1, C_, 4 * C_);
  cvt_transpose_kernel<<<(589824 + 255) / 256, 256, 0, stream>>>(fc2_w, wf2, 4 * C_, C_);

  // 2) LN1 + shift + window partition -> h16 [100352,384] f16
  ln_kernel<<<NTOK / 8, 256, 0, stream>>>(x, ln1_g, ln1_b, h16, 0);

  // 3) QKV GEMM (100352 x 1152 x 384), scatter into padded q/k/vt tiles
  gemm_wmma<0><<<dim3(1152 / 128, NTOK / 128), 256, 0, stream>>>(
      h16, wq16, qkv_b, nullptr, nullptr, q16, k16, vt16, NTOK, 1152, C_);

  // 4) attention per (window, head)
  attn_kernel<<<NWIN * NH_, 128, 0, stream>>>(q16, k16, vt16, bias_table, o16);

  // 5) proj GEMM + window-reverse + unshift + residual -> x2
  gemm_wmma<1><<<dim3(C_ / 128, NTOK / 128), 256, 0, stream>>>(
      o16, wp16, proj_b, x, x2, nullptr, nullptr, nullptr, NTOK, C_, C_);

  // 6) LN2 -> h2 f16
  ln_kernel<<<NTOK / 8, 256, 0, stream>>>(x2, ln2_g, ln2_b, h2, 1);

  // 7) fc1 GEMM + GELU -> g16 f16 (100352 x 1536 x 384)
  gemm_wmma<2><<<dim3(1536 / 128, NTOK / 128), 256, 0, stream>>>(
      h2, wf1, fc1_b, nullptr, nullptr, g16, nullptr, nullptr, NTOK, 1536, C_);

  // 8) fc2 GEMM + residual -> out f32 (100352 x 384 x 1536)
  gemm_wmma<3><<<dim3(C_ / 128, NTOK / 128), 256, 0, stream>>>(
      g16, wf2, fc2_b, x2, out, nullptr, nullptr, nullptr, NTOK, C_, 1536);
}
